// ContextualAttention_987842478426
// MI455X (gfx1250) — matrix operand order repaired and checked by reading, not compiled, not run on hardware
//
#include <hip/hip_runtime.h>
#include <hip/hip_bf16.h>

// ---------------------------------------------------------------------------
// Contextual attention for MI455X (gfx1250, wave32, WMMA bf16 16x16x32).
//
// Pipeline per batch (B=2, C=128, H=W=128, Hs=Ws=64, L=P=4096):
//   X  (4096 x 1152)  bf16 : 3x3xC patches of f_s = f[::2,::2], zero padded
//   invn(4096)        f32  : 1/max(||X_row||, 1e-4)
//   Wt (2048 x 4096)  bf16 : raw 4x4xC stride-2 patches of f, row=(c,kx,ky)
//   S  (4096 x 4096)  f32  : X @ X^T            (WMMA GEMM 1)
//   aT (4096 x 4096)  bf16 : softmax_l(10*invn[l]*S[l,p]) stored transposed
//   Z  (4096 x 2048)  f32  : aT @ Wt^T          (WMMA GEMM 2)
//   y  = overlap-add of Z patches (stride 2, kernel 4, pad 1) / 4
//
// GEMM inner loop: double-buffered LDS fed by GLOBAL_LOAD_ASYNC_TO_LDS_B128
// (ASYNCcnt), one s_wait_asynccnt + one barrier per K-step.
// ---------------------------------------------------------------------------

typedef __attribute__((ext_vector_type(16))) __bf16 v16bf;
typedef __attribute__((ext_vector_type(8)))  __bf16 v8bf;
typedef __attribute__((ext_vector_type(8)))  float  v8f;

__device__ __forceinline__ unsigned short f2bf(float x) {
  union { float f; unsigned u; } a; a.f = x;
  unsigned u = a.u;
  u += 0x7fffu + ((u >> 16) & 1u);   // round to nearest even
  return (unsigned short)(u >> 16);
}
__device__ __forceinline__ float bf2f(unsigned short h) {
  union { unsigned u; float f; } a; a.u = ((unsigned)h) << 16;
  return a.f;
}

// Async global->LDS copy, 16 bytes per lane.  GV mode: 64-bit vaddr, no saddr.
// LDS offset = low 32 bits of the generic pointer (aperture layout: addr[31:0]
// is the LDS byte offset).  Tracked by ASYNCcnt.
__device__ __forceinline__ void async_ld16(const unsigned short* g, const unsigned short* l) {
  unsigned long long ga = (unsigned long long)g;
  unsigned la = (unsigned)(unsigned long long)l;
  asm volatile("global_load_async_to_lds_b128 %0, %1, off"
               :: "v"(la), "v"(ga) : "memory");
}
__device__ __forceinline__ void wait_async0() {
  asm volatile("s_wait_asynccnt 0x0" ::: "memory");
}

// --------------------------------------------------------------------------
// X[l=(i*64+j), k=(c*9+di*3+dj)] = f_s[c, i+di-1, j+dj-1]  (0 outside)
// --------------------------------------------------------------------------
__global__ void build_x_kernel(const float* __restrict__ f, unsigned short* __restrict__ X) {
  int idx = blockIdx.x * blockDim.x + threadIdx.x;
  if (idx >= 4096 * 1152) return;
  int l = idx / 1152, k = idx - l * 1152;
  int c = k / 9, r9 = k - c * 9;
  int di = r9 / 3, dj = r9 - di * 3;
  int i = l >> 6, j = l & 63;
  int si = i + di - 1, sj = j + dj - 1;
  float v = 0.f;
  if ((unsigned)si < 64u && (unsigned)sj < 64u)
    v = f[((size_t)c * 128 + (size_t)(si * 2)) * 128 + (size_t)(sj * 2)];
  X[idx] = f2bf(v);
}

// one wave (32 lanes) per row of X -> invnorm
__global__ void rownorm_kernel(const unsigned short* __restrict__ X, float* __restrict__ invn) {
  int gid  = blockIdx.x * blockDim.x + threadIdx.x;
  int row  = gid >> 5;
  int lane = threadIdx.x & 31;
  if (row >= 4096) return;
  const unsigned short* r = X + (size_t)row * 1152;
  float s = 0.f;
  for (int k = lane; k < 1152; k += 32) { float v = bf2f(r[k]); s += v * v; }
  for (int off = 16; off; off >>= 1) s += __shfl_xor(s, off, 32);
  if (lane == 0) invn[row] = 1.0f / fmaxf(sqrtf(s), 1e-4f);
}

// Wt[m=(c*16+kx*4+ky), l=(i*64+j)] = f[c, 2i+kx-1, 2j+ky-1] (0 outside)
__global__ void build_wt_kernel(const float* __restrict__ f, unsigned short* __restrict__ Wt) {
  int idx = blockIdx.x * blockDim.x + threadIdx.x;
  if (idx >= 2048 * 4096) return;
  int m = idx >> 12, l = idx & 4095;
  int c = m >> 4, kk = m & 15;
  int kx = kk >> 2, ky = kk & 3;
  int i = l >> 6, j = l & 63;
  int si = 2 * i + kx - 1, sj = 2 * j + ky - 1;
  float v = 0.f;
  if ((unsigned)si < 128u && (unsigned)sj < 128u)
    v = f[((size_t)c * 128 + si) * 128 + sj];
  Wt[idx] = f2bf(v);
}

// --------------------------------------------------------------------------
// C(MxN,f32) = A(MxK,bf16 row-major) * B(NxK,bf16 row-major i.e. B^T)
// 128x128 block tile, K-step 32, 8 waves; each wave: 2x4 of 16x16 WMMA.
// Double-buffered LDS, async global->LDS copies, 1 barrier per K-step.
// --------------------------------------------------------------------------
__global__ __launch_bounds__(256)
void gemm_bf16_kernel(float* __restrict__ C,
                      const unsigned short* __restrict__ A,
                      const unsigned short* __restrict__ B,
                      int K, int lda, int ldb, int ldc) {
  __shared__ __align__(16) unsigned short As[2][128 * 32];
  __shared__ __align__(16) unsigned short Bs[2][128 * 32];

  const int rowBase = blockIdx.y * 128;
  const int colBase = blockIdx.x * 128;
  const int tid  = threadIdx.x;
  const int lane = tid & 31;
  const int w    = tid >> 5;        // wave 0..7
  const int wr   = w >> 1;          // 0..3 : rows [wr*32, +32)
  const int wc   = w & 1;           // 0..1 : cols [wc*64, +64)
  const int mlo  = lane & 15;
  const int half = (lane >> 4) & 1;

  // This thread's four 16B staging chunks (rows r and r+64 of each tile).
  const int r  = tid >> 2;          // 0..63
  const int cc = (tid & 3) << 3;    // 0,8,16,24 (elements)
  const unsigned short* gA0 = A + (size_t)(rowBase + r) * lda + cc;
  const unsigned short* gA1 = gA0 + (size_t)64 * lda;
  const unsigned short* gB0 = B + (size_t)(colBase + r) * ldb + cc;
  const unsigned short* gB1 = gB0 + (size_t)64 * ldb;
  const int lo0 = r * 32 + cc;
  const int lo1 = (r + 64) * 32 + cc;

  v8f acc[2][4];
#pragma unroll
  for (int i = 0; i < 2; ++i)
#pragma unroll
    for (int j = 0; j < 4; ++j)
      acc[i][j] = (v8f){0.f, 0.f, 0.f, 0.f, 0.f, 0.f, 0.f, 0.f};

  // prologue: kick off tile 0 into buffer 0
  async_ld16(gA0, &As[0][lo0]);
  async_ld16(gA1, &As[0][lo1]);
  async_ld16(gB0, &Bs[0][lo0]);
  async_ld16(gB1, &Bs[0][lo1]);

  int cur = 0;
  for (int kb = 0; kb < K; kb += 32) {
    wait_async0();        // this wave's copies for buf[cur] have landed
    __syncthreads();      // everyone's copies landed; prev reads of buf[cur^1] done

    if (kb + 32 < K) {    // prefetch next tile into the other buffer
      int nxt = cur ^ 1;
      async_ld16(gA0 + kb + 32, &As[nxt][lo0]);
      async_ld16(gA1 + kb + 32, &As[nxt][lo1]);
      async_ld16(gB0 + kb + 32, &Bs[nxt][lo0]);
      async_ld16(gB1 + kb + 32, &Bs[nxt][lo1]);
    }

    // A fragments: lane half 0 -> K 0..15, half 1 -> K 16..31
    v16bf af[2];
#pragma unroll
    for (int i = 0; i < 2; ++i) {
      int rr = wr * 32 + i * 16 + mlo;
      v8bf lo = *(const v8bf*)(&As[cur][rr * 32 + half * 16]);
      v8bf hi = *(const v8bf*)(&As[cur][rr * 32 + half * 16 + 8]);
#pragma unroll
      for (int e = 0; e < 8; ++e) { af[i][e] = lo[e]; af[i][e + 8] = hi[e]; }
    }
    // B fragments: lane half 0 -> K 0..15, half 1 -> K 16..31 (contiguous 32B)
    v16bf bf[4];
#pragma unroll
    for (int j = 0; j < 4; ++j) {
      int cidx = wc * 64 + j * 16 + mlo;
      bf[j] = *(const v16bf*)(&Bs[cur][cidx * 32 + half * 16]);
    }
#pragma unroll
    for (int i = 0; i < 2; ++i)
#pragma unroll
      for (int j = 0; j < 4; ++j)
        acc[i][j] = __builtin_amdgcn_wmma_f32_16x16x32_bf16(
            false, af[i], false, bf[j], (short)0, acc[i][j], false, false);

    cur ^= 1;
  }

  // C layout: VGPR e -> M = e + 8*half ; N = mlo
#pragma unroll
  for (int i = 0; i < 2; ++i) {
    int r0 = rowBase + wr * 32 + i * 16 + half * 8;
#pragma unroll
    for (int j = 0; j < 4; ++j) {
      int c0 = colBase + wc * 64 + j * 16 + mlo;
#pragma unroll
      for (int e = 0; e < 8; ++e)
        C[(size_t)(r0 + e) * ldc + c0] = acc[i][j][e];
    }
  }
}

// --------------------------------------------------------------------------
// Column softmax of logits v[l,p] = 10*invn[l]*S[l,p] over l; write bf16
// transposed: aT[p,l].  One thread per column p; 64 threads per block.
// --------------------------------------------------------------------------
__global__ __launch_bounds__(64)
void softmax_t_kernel(const float* __restrict__ S, const float* __restrict__ invn,
                      unsigned short* __restrict__ aT) {
  const int p = blockIdx.x * 64 + threadIdx.x;
  const int L = 4096, P = 4096;

  float m = -1e30f, s = 0.f;
  for (int l = 0; l < L; ++l) {
    float v  = 10.0f * invn[l] * S[(size_t)l * P + p];
    float nm = fmaxf(m, v);
    s = s * __expf(m - nm) + __expf(v - nm);
    m = nm;
  }
  float inv_s = 1.0f / s;

  unsigned short* dstRow = aT + (size_t)p * L;
  for (int lb = 0; lb < L; lb += 8) {
    unsigned wq[4];
#pragma unroll
    for (int q = 0; q < 4; ++q) {
      int l0 = lb + 2 * q;
      float v0 = 10.0f * invn[l0]     * S[(size_t)l0 * P + p];
      float v1 = 10.0f * invn[l0 + 1] * S[(size_t)(l0 + 1) * P + p];
      unsigned short b0 = f2bf(__expf(v0 - m) * inv_s);
      unsigned short b1 = f2bf(__expf(v1 - m) * inv_s);
      wq[q] = (unsigned)b0 | ((unsigned)b1 << 16);
    }
    uint4 pk; pk.x = wq[0]; pk.y = wq[1]; pk.z = wq[2]; pk.w = wq[3];
    *(uint4*)(dstRow + lb) = pk;
  }
}

// --------------------------------------------------------------------------
// y[c,u,v] = 0.25 * sum over (i,kx):2i+kx-1=u, (j,ky):2j+ky-1=v of
//            Z[(i*64+j), c*16+kx*4+ky]
// --------------------------------------------------------------------------
__global__ void overlap_add_kernel(const float* __restrict__ Z, float* __restrict__ y) {
  int idx = blockIdx.x * blockDim.x + threadIdx.x;
  if (idx >= 128 * 128 * 128) return;
  int v = idx & 127, u = (idx >> 7) & 127, c = idx >> 14;
  float acc = 0.f;
#pragma unroll
  for (int a = 0; a < 2; ++a) {
    int kx = ((u + 1) & 1) + 2 * a;
    int i  = (u + 1 - kx) >> 1;
    if ((unsigned)i >= 64u) continue;
#pragma unroll
    for (int bq = 0; bq < 2; ++bq) {
      int ky = ((v + 1) & 1) + 2 * bq;
      int j  = (v + 1 - ky) >> 1;
      if ((unsigned)j >= 64u) continue;
      acc += Z[(size_t)(i * 64 + j) * 2048 + (c << 4) + (kx << 2) + ky];
    }
  }
  y[idx] = 0.25f * acc;
}

// --------------------------------------------------------------------------
extern "C" void kernel_launch(void* const* d_in, const int* in_sizes, int n_in,
                              void* d_out, int out_size, void* d_ws, size_t ws_size,
                              hipStream_t stream) {
  (void)in_sizes; (void)n_in; (void)out_size; (void)ws_size;
  const float* f = (const float*)d_in[0];
  float* y = (float*)d_out;

  char* ws = (char*)d_ws;
  size_t off = 0;
  auto alloc = [&](size_t bytes) -> void* {
    void* p = ws + off;
    off += (bytes + 255) & ~(size_t)255;
    return p;
  };
  unsigned short* X   = (unsigned short*)alloc((size_t)4096 * 1152 * 2);
  float*          inv = (float*)         alloc((size_t)4096 * 4);
  unsigned short* Wt  = (unsigned short*)alloc((size_t)2048 * 4096 * 2);
  float*          S   = (float*)         alloc((size_t)4096 * 4096 * 4);
  unsigned short* aT  = (unsigned short*)alloc((size_t)4096 * 4096 * 2);
  float*          Z   = (float*)         alloc((size_t)4096 * 2048 * 4);

  for (int b = 0; b < 2; ++b) {
    const float* fb = f + (size_t)b * 128 * 128 * 128;
    float*       yb = y + (size_t)b * 128 * 128 * 128;

    { int n = 4096 * 1152;
      build_x_kernel<<<(n + 255) / 256, 256, 0, stream>>>(fb, X); }
    { int n = 4096 * 32;
      rownorm_kernel<<<(n + 255) / 256, 256, 0, stream>>>(X, inv); }
    { int n = 2048 * 4096;
      build_wt_kernel<<<(n + 255) / 256, 256, 0, stream>>>(fb, Wt); }

    { dim3 g(4096 / 128, 4096 / 128);   // S = X @ X^T
      gemm_bf16_kernel<<<g, 256, 0, stream>>>(S, X, X, 1152, 1152, 1152, 4096); }

    softmax_t_kernel<<<4096 / 64, 64, 0, stream>>>(S, inv, aT);

    { dim3 g(2048 / 128, 4096 / 128);   // Z = aT @ Wt^T
      gemm_bf16_kernel<<<g, 256, 0, stream>>>(Z, aT, Wt, 4096, 4096, 4096, 2048); }

    { int n = 128 * 128 * 128;
      overlap_add_kernel<<<(n + 255) / 256, 256, 0, stream>>>(Z, yb); }
  }
}
